// SelectedUnitsHead_85899345920644
// MI455X (gfx1250) — matrix operand order repaired and checked by reading, not compiled, not run on hardware
//
#include <hip/hip_runtime.h>
#include <hip/hip_bf16.h>

#define B_      256
#define N_      512
#define S_      64
#define E_DIM_  256
#define KEY_DIM 32
#define IN_DIM_ 1024
#define FUNC_D  256
#define UT_DIM_ 259
#define HID_    32
#define NP1     513
#define NEGV    (-1000000000.0f)

typedef __attribute__((ext_vector_type(16))) __bf16 v16bf;
typedef __attribute__((ext_vector_type(8)))  float  v8f;

// ---- WMMA fragment loaders (bf16 16x16x32, wave32) ----
// A (16x32, MxK), row-major source with leading dim ld.
__device__ __forceinline__ v16bf load_a_bf16(const float* __restrict__ base, int ld, int lane, int k0) {
  const int m  = lane & 15;
  const int lo = (lane < 16) ? 0 : 16;
  const float* row = base + (size_t)m * ld + k0;
  v16bf a;
#pragma unroll
  for (int j = 0; j < 16; ++j) {
    int k = lo + j;
    a[j] = (__bf16)row[k];
  }
  return a;
}
// B (32x16, KxN) tile of W^T where W is (OUT x K) row-major; cols [c0,c0+16), K [k0,k0+32)
__device__ __forceinline__ v16bf load_bT_bf16(const float* __restrict__ W, int ld, int lane, int k0, int c0) {
  const int n  = lane & 15;
  const int lo = (lane < 16) ? 0 : 16;
  const float* p = W + (size_t)(c0 + n) * ld + k0 + lo;
  v16bf b;
#pragma unroll
  for (int j = 0; j < 16; ++j) b[j] = (__bf16)p[j];
  return b;
}

// ---- K1: key = ent(131072x256) @ key_fc_w^T(256x32) + b ----
__global__ void key_gemm_kernel(const float* __restrict__ ent, const float* __restrict__ W,
                                const float* __restrict__ bias, float* __restrict__ key) {
  const int lane = threadIdx.x;
  const int rowBase = blockIdx.x * 16;
  v8f c0 = {}; v8f c1 = {};
  const float* abase = ent + (size_t)rowBase * E_DIM_;
#pragma unroll
  for (int k0 = 0; k0 < E_DIM_; k0 += 32) {
    v16bf a  = load_a_bf16(abase, E_DIM_, lane, k0);
    v16bf b0 = load_bT_bf16(W, E_DIM_, lane, k0, 0);
    v16bf b1 = load_bT_bf16(W, E_DIM_, lane, k0, 16);
    c0 = __builtin_amdgcn_wmma_f32_16x16x32_bf16(false, a, false, b0, (short)0, c0, false, false);
    c1 = __builtin_amdgcn_wmma_f32_16x16x32_bf16(false, a, false, b1, (short)0, c1, false, false);
  }
  const int col  = lane & 15;
  const int mofs = (lane < 16) ? 0 : 8;
#pragma unroll
  for (int r = 0; r < 8; ++r) {
    int row = rowBase + r + mofs;
    int b = row >> 9, n = row & 511;
    float* dst = key + ((size_t)b * NP1 + n) * KEY_DIM;
    dst[col]      = c0[r] + bias[col];
    dst[col + 16] = c1[r] + bias[col + 16];
  }
}

// ---- K1b: zero row N, write end_embedding at row en[b] ----
__global__ void key_fixup_kernel(const int* __restrict__ entity_num, const float* __restrict__ end_emb,
                                 float* __restrict__ key) {
  int b = blockIdx.x, k = threadIdx.x;
  key[((size_t)b * NP1 + N_) * KEY_DIM + k] = 0.0f;
  int en = entity_num[b];
  key[((size_t)b * NP1 + en) * KEY_DIM + k] = end_emb[k];
}

// ---- K2: func_embed = relu(avail(B,259) @ func_fc_w^T(259,256) + b) ----
__global__ void func_embed_kernel(const float* __restrict__ avail, const float* __restrict__ W,
                                  const float* __restrict__ bias, float* __restrict__ out) {
  int idx = blockIdx.x * blockDim.x + threadIdx.x;
  int b = idx >> 8, f = idx & 255;
  const float* a = avail + (size_t)b * UT_DIM_;
  const float* w = W + (size_t)f * UT_DIM_;
  float s = bias[f];
  for (int u = 0; u < UT_DIM_; ++u) s = fmaf(a[u], w[u], s);
  out[idx] = fmaxf(s, 0.0f);
}

// ---- K3: fused gather + embed_fc + masked cumsum -> ae_seq, new_ae ----
__global__ void gather_cumsum_kernel(const float* __restrict__ key, const int* __restrict__ sel,
                                     const int* __restrict__ sun, const int* __restrict__ entity_num,
                                     const float* __restrict__ eW, const float* __restrict__ eb,
                                     const float* __restrict__ emb, float* __restrict__ ae_seq,
                                     float* __restrict__ new_ae) {
  __shared__ float keyS[63][KEY_DIM];
  int b = blockIdx.x, tid = threadIdx.x;
  float inv_en = 1.0f / (float)entity_num[b];
  for (int e = tid; e < 63 * KEY_DIM; e += 256) {
    int s = e >> 5, k = e & 31;
    int idx = sel[b * S_ + s];
    keyS[s][k] = key[((size_t)b * NP1 + idx) * KEY_DIM + k] * inv_en;
  }
  __syncthreads();
  int sn = sun[b];
  float run[4];
#pragma unroll
  for (int g = 0; g < 4; ++g) {
    int i = tid + g * 256;
    run[g] = emb[(size_t)b * IN_DIM_ + i];
    ae_seq[(size_t)b * S_ * IN_DIM_ + i] = run[g];
  }
  for (int s = 0; s < 63; ++s) {
    float m = (s + 1 < sn) ? 1.0f : 0.0f;
#pragma unroll
    for (int g = 0; g < 4; ++g) {
      int i = tid + g * 256;
      const float* w = eW + (size_t)i * KEY_DIM;
      float d = eb[i];
#pragma unroll
      for (int k = 0; k < KEY_DIM; ++k) d = fmaf(keyS[s][k], w[k], d);
      run[g] += d * m;
      ae_seq[((size_t)b * S_ + (s + 1)) * IN_DIM_ + i] = run[g];
    }
  }
#pragma unroll
  for (int g = 0; g < 4; ++g) new_ae[(size_t)b * IN_DIM_ + tid + g * 256] = run[g];
}

// ---- K4a: q_in = relu(ae_seq(16384x1024) @ fc1_w^T + b + func_embed) ----
// 4 column tiles (64 cols) per block: A fragment loaded once per K-step,
// 4 independent WMMA chains hide the bf16 WMMA->WMMA RAW hazard window.
__global__ void fc1_kernel(const float* __restrict__ ae_seq, const float* __restrict__ W,
                           const float* __restrict__ bias, const float* __restrict__ func_embed,
                           float* __restrict__ q_in) {
  const int lane = threadIdx.x;
  const int rowBase = blockIdx.x * 16;
  const int colBase = blockIdx.y * 64;
  v8f acc[4] = {};
  const float* abase = ae_seq + (size_t)rowBase * IN_DIM_;
  for (int k0 = 0; k0 < IN_DIM_; k0 += 32) {
    v16bf a = load_a_bf16(abase, IN_DIM_, lane, k0);
#pragma unroll
    for (int t = 0; t < 4; ++t) {
      v16bf bb = load_bT_bf16(W, IN_DIM_, lane, k0, colBase + t * 16);
      acc[t] = __builtin_amdgcn_wmma_f32_16x16x32_bf16(false, a, false, bb, (short)0, acc[t], false, false);
    }
  }
  const int mofs = (lane < 16) ? 0 : 8;
#pragma unroll
  for (int t = 0; t < 4; ++t) {
    const int f = colBase + t * 16 + (lane & 15);
#pragma unroll
    for (int r = 0; r < 8; ++r) {
      int row = rowBase + r + mofs;
      int b = row >> 6;
      float v = acc[t][r] + bias[f] + func_embed[(size_t)b * FUNC_D + f];
      q_in[(size_t)row * FUNC_D + f] = fmaxf(v, 0.0f);
    }
  }
}

// ---- K4b: x_seq = q_in(16384x256) @ fc2_w^T(256x32) + b ----
__global__ void fc2_kernel(const float* __restrict__ q_in, const float* __restrict__ W,
                           const float* __restrict__ bias, float* __restrict__ x_seq) {
  const int lane = threadIdx.x;
  const int rowBase = blockIdx.x * 16;
  v8f c0 = {}; v8f c1 = {};
  const float* abase = q_in + (size_t)rowBase * FUNC_D;
#pragma unroll
  for (int k0 = 0; k0 < FUNC_D; k0 += 32) {
    v16bf a  = load_a_bf16(abase, FUNC_D, lane, k0);
    v16bf b0 = load_bT_bf16(W, FUNC_D, lane, k0, 0);
    v16bf b1 = load_bT_bf16(W, FUNC_D, lane, k0, 16);
    c0 = __builtin_amdgcn_wmma_f32_16x16x32_bf16(false, a, false, b0, (short)0, c0, false, false);
    c1 = __builtin_amdgcn_wmma_f32_16x16x32_bf16(false, a, false, b1, (short)0, c1, false, false);
  }
  const int col  = lane & 15;
  const int mofs = (lane < 16) ? 0 : 8;
#pragma unroll
  for (int r = 0; r < 8; ++r) {
    int row = rowBase + r + mofs;
    x_seq[(size_t)row * HID_ + col]      = c0[r] + bias[col];
    x_seq[(size_t)row * HID_ + col + 16] = c1[r] + bias[col + 16];
  }
}

// ---- K5: LSTM over 64 steps, one block per batch ----
__device__ __forceinline__ float sigf(float x) { return 1.0f / (1.0f + __expf(-x)); }

__global__ void lstm_kernel(const float* __restrict__ x_seq, const float* __restrict__ wih,
                            const float* __restrict__ whh, const float* __restrict__ bih,
                            const float* __restrict__ bhh, float* __restrict__ queries) {
  __shared__ float wihS[4 * HID_][HID_ + 1];
  __shared__ float whhS[4 * HID_][HID_ + 1];
  __shared__ float hS[HID_], cS[HID_], gS[4 * HID_], xS[HID_];
  int b = blockIdx.x, tid = threadIdx.x; // 128 threads
  for (int i = tid; i < 4 * HID_ * HID_; i += 128) {
    wihS[i >> 5][i & 31] = wih[i];
    whhS[i >> 5][i & 31] = whh[i];
  }
  if (tid < HID_) { hS[tid] = 0.f; cS[tid] = 0.f; }
  float bsum = bih[tid] + bhh[tid];
  __syncthreads();
  for (int s = 0; s < S_; ++s) {
    if (tid < HID_) xS[tid] = x_seq[((size_t)b * S_ + s) * HID_ + tid];
    __syncthreads();
    float g = bsum;
#pragma unroll
    for (int k = 0; k < HID_; ++k) g = fmaf(xS[k], wihS[tid][k], g);
#pragma unroll
    for (int k = 0; k < HID_; ++k) g = fmaf(hS[k], whhS[tid][k], g);
    gS[tid] = g;
    __syncthreads();
    if (tid < HID_) {
      float ig = sigf(gS[tid]);
      float fg = sigf(gS[HID_ + tid]);
      float gg = tanhf(gS[2 * HID_ + tid]);
      float og = sigf(gS[3 * HID_ + tid]);
      float cn = fg * cS[tid] + ig * gg;
      float hn = og * tanhf(cn);
      cS[tid] = cn; hS[tid] = hn;
      queries[((size_t)b * S_ + s) * HID_ + tid] = hn;
    }
    __syncthreads();
  }
}

// ---- K6: logits = q @ key^T with mask (prev_sel via first-selected-step table) ----
__global__ void logits_kernel(const float* __restrict__ queries, const float* __restrict__ key,
                              const int* __restrict__ sel, const int* __restrict__ entity_num,
                              float* __restrict__ out) {
  __shared__ float qS[S_ * KEY_DIM];
  __shared__ int firstSel[NP1];
  int b = blockIdx.x, tid = threadIdx.x;
  for (int i = tid; i < S_ * KEY_DIM; i += 256) qS[i] = queries[(size_t)b * S_ * KEY_DIM + i];
  for (int i = tid; i < NP1; i += 256) firstSel[i] = 1 << 30;
  __syncthreads();
  if (tid < S_) atomicMin(&firstSel[sel[b * S_ + tid]], tid);
  __syncthreads();
  int en = entity_num[b];
  const float* kb = key + (size_t)b * NP1 * KEY_DIM;
  for (int idx = tid; idx < S_ * NP1; idx += 256) {
    int s = idx / NP1;
    int n = idx - s * NP1;
    float v = NEGV;
    bool valid = (n <= en) && (firstSel[n] >= s) && !((n == en) && (s == 0));
    if (valid) {
      const float* kr = kb + (size_t)n * KEY_DIM;
      const float* qr = qS + s * KEY_DIM;
      float d = 0.f;
#pragma unroll
      for (int k = 0; k < KEY_DIM; ++k) d = fmaf(qr[k], kr[k], d);
      v = d;
    }
    out[((size_t)b * S_ + s) * NP1 + n] = v;
  }
}

extern "C" void kernel_launch(void* const* d_in, const int* in_sizes, int n_in,
                              void* d_out, int out_size, void* d_ws, size_t ws_size,
                              hipStream_t stream) {
  const float* embedding  = (const float*)d_in[0];
  const float* avail      = (const float*)d_in[1];
  const float* ent        = (const float*)d_in[2];
  const int*   entity_num = (const int*)d_in[3];
  const int*   sel        = (const int*)d_in[4];
  const int*   sun        = (const int*)d_in[5];
  const float* key_fc_w   = (const float*)d_in[6];
  const float* key_fc_b   = (const float*)d_in[7];
  const float* func_fc_w  = (const float*)d_in[8];
  const float* func_fc_b  = (const float*)d_in[9];
  const float* fc1_w      = (const float*)d_in[10];
  const float* fc1_b      = (const float*)d_in[11];
  const float* fc2_w      = (const float*)d_in[12];
  const float* fc2_b      = (const float*)d_in[13];
  const float* embed_fc_w = (const float*)d_in[14];
  const float* embed_fc_b = (const float*)d_in[15];
  const float* end_emb    = (const float*)d_in[16];
  const float* lstm_wih   = (const float*)d_in[17];
  const float* lstm_bih   = (const float*)d_in[18];
  const float* lstm_whh   = (const float*)d_in[19];
  const float* lstm_bhh   = (const float*)d_in[20];

  float* ws = (float*)d_ws;
  size_t o = 0;
  float* key      = ws + o; o += (size_t)B_ * NP1 * KEY_DIM;     // 4.20M
  float* func_emb = ws + o; o += (size_t)B_ * FUNC_D;            // 64K
  float* ae_seq   = ws + o; o += (size_t)B_ * S_ * IN_DIM_;      // 16.8M
  float* q_in     = ws + o; o += (size_t)B_ * S_ * FUNC_D;       // 4.19M
  float* x_seq    = ws + o; o += (size_t)B_ * S_ * HID_;         // 0.52M
  float* queries  = ws + o; o += (size_t)B_ * S_ * HID_;         // 0.52M

  float* out_logits = (float*)d_out;
  float* out_ae     = out_logits + (size_t)B_ * S_ * NP1;

  hipLaunchKernelGGL(key_gemm_kernel, dim3((B_ * N_) / 16), dim3(32), 0, stream,
                     ent, key_fc_w, key_fc_b, key);
  hipLaunchKernelGGL(key_fixup_kernel, dim3(B_), dim3(KEY_DIM), 0, stream,
                     entity_num, end_emb, key);
  hipLaunchKernelGGL(func_embed_kernel, dim3(B_), dim3(256), 0, stream,
                     avail, func_fc_w, func_fc_b, func_emb);
  hipLaunchKernelGGL(gather_cumsum_kernel, dim3(B_), dim3(256), 0, stream,
                     key, sel, sun, entity_num, embed_fc_w, embed_fc_b, embedding, ae_seq, out_ae);
  hipLaunchKernelGGL(fc1_kernel, dim3((B_ * S_) / 16, FUNC_D / 64), dim3(32), 0, stream,
                     ae_seq, fc1_w, fc1_b, func_emb, q_in);
  hipLaunchKernelGGL(fc2_kernel, dim3((B_ * S_) / 16), dim3(32), 0, stream,
                     q_in, fc2_w, fc2_b, x_seq);
  hipLaunchKernelGGL(lstm_kernel, dim3(B_), dim3(4 * HID_), 0, stream,
                     x_seq, lstm_wih, lstm_whh, lstm_bih, lstm_bhh, queries);
  hipLaunchKernelGGL(logits_kernel, dim3(B_), dim3(256), 0, stream,
                     queries, key, sel, entity_num, out_logits);
}
